// RelativeAttention_6313601925598
// MI455X (gfx1250) — matrix operand order repaired and checked by reading, not compiled, not run on hardware
//
#include <hip/hip_runtime.h>
#include <math.h>

typedef __attribute__((ext_vector_type(16))) __bf16  v16bf;
typedef __attribute__((ext_vector_type(8)))  __bf16  v8bf;
typedef __attribute__((ext_vector_type(4)))  __bf16  v4bf;
typedef __attribute__((ext_vector_type(8)))  float   v8f;
typedef __attribute__((ext_vector_type(4)))  float   f32x4;

#define DIM   768
#define NH    12
#define HD    64
#define BATCH 8
#define SEQ   1024
#define LDK   72   // LDS row stride (elements): 16B-aligned rows, breaks bank-stride

#define WAIT_ASYNC(n) asm volatile("s_wait_asynccnt " #n ::: "memory")

static __device__ __forceinline__ void async_copy16(const __bf16* g, __bf16* l) {
  unsigned lds = (unsigned)(unsigned long long)(void*)l;       // LDS offset (low 32 bits)
  unsigned long long ga = (unsigned long long)g;
  asm volatile("global_load_async_to_lds_b128 %0, %1, off" :: "v"(lds), "v"(ga) : "memory");
}

// Copy a 64-row x 64-col bf16 tile (global row stride ldg) into LDS [row][LDK]
// as 512 x 16B async chunks; 128 threads -> 4 chunks each.
static __device__ __forceinline__ void stage_tile(const __bf16* gsrc, int ldg,
                                                  __bf16* lbase, int tid) {
#pragma unroll
  for (int j = 0; j < 4; j++) {
    int q = tid + j * 128;
    int row = q >> 3, ck = q & 7;
    async_copy16(gsrc + (size_t)row * ldg + ck * 8, lbase + row * LDK + ck * 8);
  }
}

static __device__ __forceinline__ v16bf catbf(v8bf lo, v8bf hi) {
  return __builtin_shufflevector(lo, hi, 0, 1, 2, 3, 4, 5, 6, 7, 8, 9, 10, 11, 12, 13, 14, 15);
}

// A-fragment (16x32 bf16) from a bf16 row pointer (global or LDS); lane row M = lane&15.
// Lane layout: e0..7 -> K = k0 + 8*hk + e ; e8..15 -> K = k0 + 16 + 8*hk + (e-8)
static __device__ __forceinline__ v16bf a_frag_bf16(const __bf16* rowp, int k0, int hk) {
  const __bf16* p0 = rowp + k0 + hk * 8;
  v8bf lo = *(const v8bf*)(p0);
  v8bf hi = *(const v8bf*)(p0 + 16);
  return catbf(lo, hi);
}

// B-fragment (32x16 bf16) from an LDS tile stored [N][K] with row stride ld.
// Lane layout: N = lane&15 ; e0..7 -> K = k0 + 8*(lane>>4) + e ; e8..15 -> K = k0 + 16 + 8*(lane>>4) + (e-8)
static __device__ __forceinline__ v16bf b_frag_nk(const __bf16* tile, int ld,
                                                  int n0, int k0, int lane) {
  const __bf16* p = tile + (n0 + (lane & 15)) * ld + k0 + ((lane >> 4) << 3);
  v8bf lo = *(const v8bf*)(p);
  v8bf hi = *(const v8bf*)(p + 16);
  return catbf(lo, hi);
}

static __device__ __forceinline__ v8f wmma_bf16(v16bf a, v16bf b, v8f c) {
  return __builtin_amdgcn_wmma_f32_16x16x32_bf16(false, a, false, b, (short)0, c, false, false);
}

static __device__ __forceinline__ float redmax16(float v) {
#pragma unroll
  for (int m = 1; m < 16; m <<= 1) v = fmaxf(v, __shfl_xor(v, m, 32));
  return v;
}
static __device__ __forceinline__ float redsum16(float v) {
#pragma unroll
  for (int m = 1; m < 16; m <<= 1) v += __shfl_xor(v, m, 32);
  return v;
}

// ---------------- Kernel X: convert x to bf16 (one pass) ----------------
__global__ __launch_bounds__(256) void conv_x(const float* __restrict__ xf,
                                              __bf16* __restrict__ xb) {
  size_t i = ((size_t)blockIdx.x * 256 + threadIdx.x) * 4;
  f32x4 v = *(const f32x4*)(xf + i);
  *(v4bf*)(xb + i) = __builtin_convertvector(v, v4bf);
}

// ---------------- Kernel W: transpose+convert weights to bf16 [C][K] ----------------
__global__ __launch_bounds__(256) void conv_wT(const float* __restrict__ w,
                                               __bf16* __restrict__ wT, int C) {
  size_t i = (size_t)blockIdx.x * 256 + threadIdx.x;   // i = c*768 + k
  if (i >= (size_t)C * DIM) return;
  int c = (int)(i / DIM);
  int k = (int)(i - (size_t)c * DIM);
  wT[i] = (__bf16)w[(size_t)k * C + c];
}

// ---------------- Kernel 0: expand relative bias to [H][N][M] bf16 ----------------
__global__ __launch_bounds__(256) void bias_expand(const float* __restrict__ bias_table,
                                                   const int* __restrict__ rel_index,
                                                   __bf16* __restrict__ bias_full) {
  const int n = blockIdx.x;
  for (int m = threadIdx.x; m < SEQ; m += 256) {
    int idx = rel_index[(n << 10) + m];
    const float* tp = bias_table + idx * NH;
#pragma unroll
    for (int h = 0; h < NH; h++)
      bias_full[((size_t)h << 20) + (n << 10) + m] = (__bf16)tp[h];
  }
}

// ---------------- Kernel 1: QKV projection  (xb[8192x768] @ wT[2304x768]) ----------------
// Workgroup: 128 rows x 64 cols; each wave: 32 rows (2 row-subtiles) x 64 cols.
__global__ __launch_bounds__(128) void qkv_gemm(const __bf16* __restrict__ xb,
                                                const __bf16* __restrict__ wT,
                                                __bf16* __restrict__ qb,
                                                __bf16* __restrict__ kb,
                                                __bf16* __restrict__ vb) {
  __shared__ __bf16 ldsB[2][64 * LDK];
  const int tid = threadIdx.x;
  const int lane = tid & 31, w = tid >> 5;
  const int hk = lane >> 4, ln = lane & 15;
  const int c0 = blockIdx.x * 64;
  const int m0 = blockIdx.y * 128;
  const __bf16* xrow0 = xb + (size_t)(m0 + w * 32 + ln) * DIM;
  const __bf16* xrow1 = xrow0 + (size_t)16 * DIM;
  const __bf16* wblk = wT + (size_t)c0 * DIM;

  v8f z = {0.f, 0.f, 0.f, 0.f, 0.f, 0.f, 0.f, 0.f};
  v8f acc[2][4];
#pragma unroll
  for (int u = 0; u < 2; u++)
#pragma unroll
    for (int t = 0; t < 4; t++) acc[u][t] = z;

  stage_tile(wblk, DIM, ldsB[0], tid);              // prologue: async stage K0=0
#pragma unroll 1
  for (int kt = 0; kt < 12; kt++) {
    __bf16* cur = ldsB[kt & 1];
    __syncthreads();                                // no wave still reads 'nxt'
    if (kt < 11) {
      stage_tile(wblk + (kt + 1) * 64, DIM, ldsB[(kt + 1) & 1], tid);
      WAIT_ASYNC(4);                                // current tile (first 4) complete
    } else {
      WAIT_ASYNC(0);
    }
    __syncthreads();                                // whole WG sees current tile
    int K0 = kt * 64;
#pragma unroll
    for (int ka = 0; ka < 2; ka++) {
      v16bf a0 = a_frag_bf16(xrow0, K0 + 32 * ka, hk);
      v16bf a1 = a_frag_bf16(xrow1, K0 + 32 * ka, hk);
#pragma unroll
      for (int t = 0; t < 4; t++) {
        v16bf bf = b_frag_nk(cur, LDK, 16 * t, 32 * ka, lane);
        acc[0][t] = wmma_bf16(a0, bf, acc[0][t]);
        acc[1][t] = wmma_bf16(a1, bf, acc[1][t]);
      }
    }
  }

#pragma unroll
  for (int u = 0; u < 2; u++) {
#pragma unroll
    for (int t = 0; t < 4; t++) {
      int cbase = c0 + 16 * t;
      int s = cbase / DIM;               // 0=q 1=k 2=v (16-wide tile never straddles)
      int within = cbase - s * DIM;
      int h = within >> 6;
      int dbase = within & 63;
#pragma unroll
      for (int r = 0; r < 8; r++) {
        int mrow = m0 + w * 32 + u * 16 + r + 8 * hk;
        int b_ = mrow >> 10, n_ = mrow & 1023;
        size_t bh = (size_t)(b_ * NH + h);
        __bf16 val = (__bf16)acc[u][t][r];
        if (s == 0)      qb[(((bh << 10) + n_) << 6) + dbase + ln] = val;
        else if (s == 1) kb[(((bh << 10) + n_) << 6) + dbase + ln] = val;
        else             vb[((bh * HD + dbase + ln) << 10) + n_] = val;  // V transposed [bh][d][n]
      }
    }
  }
}

// ---------------- Kernel 2: fused attention with relative bias ----------------
__global__ __launch_bounds__(128) void attn_kernel(const __bf16* __restrict__ qb,
                                                   const __bf16* __restrict__ kb,
                                                   const __bf16* __restrict__ vb,
                                                   const __bf16* __restrict__ bias_full,
                                                   __bf16* __restrict__ attn_out) {
  __shared__ __bf16 ldsK[2][64 * LDK];   // [m][hd]  (B-as-[N][K] for Q*K^T)
  __shared__ __bf16 ldsV[2][64 * LDK];   // [hd][m]  (B-as-[N][K] for P*V)
  __shared__ __bf16 ldsP[4 * 16 * LDK];
  const int tid = threadIdx.x, lane = tid & 31, w = tid >> 5;
  const int hk = lane >> 4, ln = lane & 15;
  const int bh = blockIdx.y;            // b*NH + h
  const int h = bh % NH, b = bh / NH;
  const int n0 = blockIdx.x * 64 + w * 16;
  const __bf16* kbase = kb + ((size_t)bh << 16);          // [1024][64]
  const __bf16* vbase = vb + ((size_t)bh << 16);          // [64][1024]

  const __bf16* qrow = qb + ((size_t)((bh << 10) + n0 + ln)) * HD;
  v16bf qa0 = a_frag_bf16(qrow, 0, hk);
  v16bf qa1 = a_frag_bf16(qrow, 32, hk);
  const float scale = 0.125f;           // 64^-0.5
  const __bf16* hbias = bias_full + ((size_t)h << 20);

  v8f z = {0.f, 0.f, 0.f, 0.f, 0.f, 0.f, 0.f, 0.f};
  v8f o[4];
#pragma unroll
  for (int t = 0; t < 4; t++) o[t] = z;
  float rmax[8], rsum[8], fac[8];
#pragma unroll
  for (int r = 0; r < 8; r++) { rmax[r] = -3.0e38f; rsum[r] = 0.f; }

  stage_tile(kbase, HD, ldsK[0], tid);                    // prologue m0=0
  stage_tile(vbase, SEQ, ldsV[0], tid);
#pragma unroll 1
  for (int it = 0; it < 16; it++) {
    const int m0 = it * 64;
    __bf16* curK = ldsK[it & 1];
    __bf16* curV = ldsV[it & 1];
    __syncthreads();                                      // prev readers of 'nxt' done
    if (it < 15) {
      stage_tile(kbase + (size_t)(m0 + 64) * HD, HD, ldsK[(it + 1) & 1], tid);
      stage_tile(vbase + (m0 + 64), SEQ, ldsV[(it + 1) & 1], tid);
      WAIT_ASYNC(8);                                      // current 8 chunks complete
    } else {
      WAIT_ASYNC(0);
    }
    __syncthreads();

    // scores S = Q*K^T (16 rows x 64 cols per wave)
    v8f s[4];
#pragma unroll
    for (int t = 0; t < 4; t++) {
      v8f a0 = wmma_bf16(qa0, b_frag_nk(curK, LDK, 16 * t, 0, lane), z);
      s[t] = wmma_bf16(qa1, b_frag_nk(curK, LDK, 16 * t, 32, lane), a0);
    }
    // scale + relative bias (coalesced bf16 loads from expanded table)
#pragma unroll
    for (int t = 0; t < 4; t++) {
#pragma unroll
      for (int r = 0; r < 8; r++) {
        int nrow = n0 + r + 8 * hk;
        int mcol = m0 + 16 * t + ln;
        float bias = (float)hbias[((size_t)nrow << 10) + mcol];
        s[t][r] = s[t][r] * scale + bias;
      }
    }
    // online softmax (rows M = r + 8*hk; reduce across the 16 lanes sharing a row)
#pragma unroll
    for (int r = 0; r < 8; r++) {
      float tm = fmaxf(fmaxf(s[0][r], s[1][r]), fmaxf(s[2][r], s[3][r]));
      tm = redmax16(tm);
      float nm = fmaxf(rmax[r], tm);
      fac[r] = __expf(rmax[r] - nm);
      rmax[r] = nm;
    }
#pragma unroll
    for (int t = 0; t < 4; t++) {
#pragma unroll
      for (int r = 0; r < 8; r++) s[t][r] = __expf(s[t][r] - rmax[r]);
    }
#pragma unroll
    for (int r = 0; r < 8; r++) {
      float ps = s[0][r] + s[1][r] + s[2][r] + s[3][r];
      ps = redsum16(ps);
      rsum[r] = rsum[r] * fac[r] + ps;
    }
#pragma unroll
    for (int t = 0; t < 4; t++) {
#pragma unroll
      for (int r = 0; r < 8; r++) o[t][r] *= fac[r];
    }
    // spill P (bf16) to wave-private LDS, re-read in A-fragment layout
    __bf16* pbase = ldsP + w * 16 * LDK;
#pragma unroll
    for (int t = 0; t < 4; t++) {
#pragma unroll
      for (int r = 0; r < 8; r++)
        pbase[(r + 8 * hk) * LDK + 16 * t + ln] = (__bf16)s[t][r];
    }
    asm volatile("s_wait_dscnt 0" ::: "memory");   // order in-wave LDS RAW
    v16bf pa0 = a_frag_bf16(pbase + ln * LDK, 0, hk);
    v16bf pa1 = a_frag_bf16(pbase + ln * LDK, 32, hk);
#pragma unroll
    for (int t = 0; t < 4; t++) {
      o[t] = wmma_bf16(pa0, b_frag_nk(curV, LDK, 16 * t, 0, lane), o[t]);
      o[t] = wmma_bf16(pa1, b_frag_nk(curV, LDK, 16 * t, 32, lane), o[t]);
    }
  }

  // normalize + store to [b][n][h*64+d] bf16
  float inv[8];
#pragma unroll
  for (int r = 0; r < 8; r++) inv[r] = 1.0f / rsum[r];
#pragma unroll
  for (int t = 0; t < 4; t++) {
#pragma unroll
    for (int r = 0; r < 8; r++) {
      int nrow = n0 + r + 8 * hk;
      attn_out[((size_t)((b << 10) + nrow)) * DIM + h * HD + 16 * t + ln] =
          (__bf16)(o[t][r] * inv[r]);
    }
  }
}

// ---------------- Kernel 3: output projection + bias ----------------
__global__ __launch_bounds__(128) void proj_gemm(const __bf16* __restrict__ ain,
                                                 const __bf16* __restrict__ wT,
                                                 const float* __restrict__ proj_b,
                                                 float* __restrict__ out) {
  __shared__ __bf16 ldsB[2][64 * LDK];
  const int tid = threadIdx.x;
  const int lane = tid & 31, w = tid >> 5;
  const int hk = lane >> 4, ln = lane & 15;
  const int c0 = blockIdx.x * 64;
  const int m0 = blockIdx.y * 128;
  const __bf16* arow0 = ain + (size_t)(m0 + w * 32 + ln) * DIM;
  const __bf16* arow1 = arow0 + (size_t)16 * DIM;
  const __bf16* wblk = wT + (size_t)c0 * DIM;

  v8f z = {0.f, 0.f, 0.f, 0.f, 0.f, 0.f, 0.f, 0.f};
  v8f acc[2][4];
#pragma unroll
  for (int u = 0; u < 2; u++)
#pragma unroll
    for (int t = 0; t < 4; t++) acc[u][t] = z;

  stage_tile(wblk, DIM, ldsB[0], tid);
#pragma unroll 1
  for (int kt = 0; kt < 12; kt++) {
    __bf16* cur = ldsB[kt & 1];
    __syncthreads();
    if (kt < 11) {
      stage_tile(wblk + (kt + 1) * 64, DIM, ldsB[(kt + 1) & 1], tid);
      WAIT_ASYNC(4);
    } else {
      WAIT_ASYNC(0);
    }
    __syncthreads();
    int K0 = kt * 64;
#pragma unroll
    for (int ka = 0; ka < 2; ka++) {
      v16bf a0 = a_frag_bf16(arow0, K0 + 32 * ka, hk);
      v16bf a1 = a_frag_bf16(arow1, K0 + 32 * ka, hk);
#pragma unroll
      for (int t = 0; t < 4; t++) {
        v16bf bf = b_frag_nk(cur, LDK, 16 * t, 32 * ka, lane);
        acc[0][t] = wmma_bf16(a0, bf, acc[0][t]);
        acc[1][t] = wmma_bf16(a1, bf, acc[1][t]);
      }
    }
  }

#pragma unroll
  for (int u = 0; u < 2; u++) {
#pragma unroll
    for (int t = 0; t < 4; t++) {
      int c = c0 + 16 * t + ln;
      float pb = proj_b[c];
#pragma unroll
      for (int r = 0; r < 8; r++) {
        int mrow = m0 + w * 32 + u * 16 + r + 8 * hk;
        out[(size_t)mrow * DIM + c] = acc[u][t][r] + pb;
      }
    }
  }
}

extern "C" void kernel_launch(void* const* d_in, const int* in_sizes, int n_in,
                              void* d_out, int out_size, void* d_ws, size_t ws_size,
                              hipStream_t stream) {
  (void)in_sizes; (void)n_in; (void)out_size; (void)ws_size;
  const float* x          = (const float*)d_in[0];
  const float* qkv_w      = (const float*)d_in[1];
  const float* proj_w     = (const float*)d_in[2];
  const float* proj_b     = (const float*)d_in[3];
  const float* bias_table = (const float*)d_in[4];
  const int*   rel_index  = (const int*)d_in[5];
  float* out = (float*)d_out;

  const size_t per = (size_t)BATCH * NH * SEQ * HD;   // elements per q/k/v buffer
  __bf16* qb  = (__bf16*)d_ws;
  __bf16* kb  = qb + per;
  __bf16* vb  = kb + per;
  __bf16* ao  = vb + per;                             // [B*SEQ][DIM] bf16
  __bf16* bfb = ao + (size_t)BATCH * SEQ * DIM;       // [NH][SEQ][SEQ] bf16 bias
  __bf16* qwT = bfb + ((size_t)NH << 20);             // [2304][768] bf16
  __bf16* pwT = qwT + (size_t)3 * DIM * DIM;          // [768][768] bf16
  __bf16* xbb = pwT + (size_t)DIM * DIM;              // [8192][768] bf16

  conv_x<<<dim3((BATCH * SEQ * DIM) / 1024), dim3(256), 0, stream>>>(x, xbb);
  conv_wT<<<dim3((3 * DIM * DIM + 255) / 256), dim3(256), 0, stream>>>(qkv_w, qwT, 3 * DIM);
  conv_wT<<<dim3((DIM * DIM + 255) / 256), dim3(256), 0, stream>>>(proj_w, pwT, DIM);
  bias_expand<<<dim3(SEQ), dim3(256), 0, stream>>>(bias_table, rel_index, bfb);
  qkv_gemm<<<dim3(36, 64), dim3(128), 0, stream>>>(xbb, qwT, qb, kb, vb);
  attn_kernel<<<dim3(SEQ / 64, BATCH * NH), dim3(128), 0, stream>>>(qb, kb, vb, bfb, ao);
  proj_gemm<<<dim3(DIM / 64, 64), dim3(128), 0, stream>>>(ao, pwT, proj_b, out);
}
